// TransitionLayer_3332894621736
// MI455X (gfx1250) — compile-verified, matrix operand order checked by reading, not run on hardware
//
#include <hip/hip_runtime.h>
#include <cmath>
#include <cfloat>

// ---------------- CDNA5 (gfx1250) types ----------------
typedef __bf16 bf16_t;
typedef __attribute__((ext_vector_type(16))) __bf16 v16bf;
typedef __attribute__((ext_vector_type(8)))  __bf16 v8bf;
typedef __attribute__((ext_vector_type(8)))  float  v8f;

#define N_CODES 12288
#define GRAPH   128
#define HIDDEN  128
#define ATT     64

// ---------------- workspace layout (bytes) ----------------
static constexpr size_t OFF_CNT   = 0;                                  // int[4]
static constexpr size_t OFF_TF    = 256;                                // float[128]
static constexpr size_t OFF_M1    = 1024;                               // float[128]
static constexpr size_t OFF_M23   = 2048;                               // float[128]
static constexpr size_t OFF_IDX23 = 4096;                               // int[N]
static constexpr size_t OFF_IDX1  = OFF_IDX23 + (size_t)N_CODES * 4;    // int[N]
// bf16 hi/lo split weights
static constexpr size_t SZ_WQK    = (size_t)ATT * GRAPH * 2;            // 16 KB
static constexpr size_t SZ_WV     = (size_t)HIDDEN * GRAPH * 2;         // 32 KB
static constexpr size_t SZ_WG     = (size_t)3 * HIDDEN * GRAPH * 2;     // 96 KB
static constexpr size_t OFF_WQH   = OFF_IDX1 + (size_t)N_CODES * 4;     // 102400 (256-aligned)
static constexpr size_t OFF_WQL   = OFF_WQH + SZ_WQK;
static constexpr size_t OFF_WKH   = OFF_WQL + SZ_WQK;
static constexpr size_t OFF_WKL   = OFF_WKH + SZ_WQK;
static constexpr size_t OFF_WVH   = OFF_WKL + SZ_WQK;
static constexpr size_t OFF_WVL   = OFF_WVH + SZ_WV;
static constexpr size_t OFF_WIH_H = OFF_WVL + SZ_WV;
static constexpr size_t OFF_WIH_L = OFF_WIH_H + SZ_WG;
static constexpr size_t OFF_WHH_H = OFF_WIH_L + SZ_WG;
static constexpr size_t OFF_WHH_L = OFF_WHH_H + SZ_WG;
// bf16 hi/lo staging for attention
static constexpr size_t SZ_QK     = (size_t)N_CODES * ATT * 2;          // 1.5 MB
static constexpr size_t SZ_VT     = (size_t)HIDDEN * N_CODES * 2;       // 3 MB
static constexpr size_t OFF_QCH   = 786432;  // past weights (~627KB), 256-aligned
static constexpr size_t OFF_QCL   = OFF_QCH + SZ_QK;
static constexpr size_t OFF_KCH   = OFF_QCL + SZ_QK;
static constexpr size_t OFF_KCL   = OFF_KCH + SZ_QK;
static constexpr size_t OFF_VTH   = OFF_KCL + SZ_QK;
static constexpr size_t OFF_VTL   = OFF_VTH + SZ_VT;
static constexpr size_t OFF_END   = OFF_VTL + SZ_VT;                    // ~13.4 MB

// ---------------- helpers ----------------
__device__ inline v8f wmma_bf16(v16bf a, v16bf b, v8f c) {
  return __builtin_amdgcn_wmma_f32_16x16x32_bf16(false, a, false, b, (short)0, c, false, false);
}
// bf16x3 emulated-fp32 product: A*B ~= AhBh + AlBh + AhBl
__device__ inline v8f mma3(v16bf ah, v16bf al, v16bf bh, v16bf bl, v8f c) {
  c = wmma_bf16(ah, bl, c);
  c = wmma_bf16(al, bh, c);
  c = wmma_bf16(ah, bh, c);
  return c;
}

// split 8 consecutive fp32 into bf16 hi/lo
__device__ inline void cvt8s(const float* __restrict__ p, __bf16* dh, __bf16* dl) {
  float4 a = *reinterpret_cast<const float4*>(p);
  float4 b = *reinterpret_cast<const float4*>(p + 4);
  float v[8] = {a.x, a.y, a.z, a.w, b.x, b.y, b.z, b.w};
#pragma unroll
  for (int i = 0; i < 8; ++i) {
    __bf16 hi = (__bf16)v[i];
    dh[i] = hi;
    dl[i] = (__bf16)(v[i] - (float)hi);
  }
}
// A-layout (16-bit, 16x32): half h: elems 0..7 -> K=kbase+8h+0..7, 8..15 -> K=kbase+16+8h+0..7
__device__ inline void loadA_split(const float* __restrict__ row, int kbase, int h,
                                   v16bf* hi, v16bf* lo) {
  __bf16* dh = reinterpret_cast<__bf16*>(hi);
  __bf16* dl = reinterpret_cast<__bf16*>(lo);
  cvt8s(row + kbase + 8 * h, dh, dl);
  cvt8s(row + kbase + 16 + 8 * h, dh + 8, dl + 8);
}
__device__ inline v16bf loadA_bf(const bf16_t* __restrict__ base, int offLo, int offHi) {
  v16bf r;
  v8bf* d = reinterpret_cast<v8bf*>(&r);
  d[0] = *reinterpret_cast<const v8bf*>(base + offLo);
  d[1] = *reinterpret_cast<const v8bf*>(base + offHi);
  return r;
}

// Order-independent (deterministic) float atomic max via CAS.
__device__ inline void atomicMaxF(float* addr, float val) {
  unsigned int* ua = reinterpret_cast<unsigned int*>(addr);
  unsigned int old = __float_as_uint(*addr);
  while (__uint_as_float(old) < val) {
    unsigned int assumed = old;
    old = atomicCAS(ua, assumed, __float_as_uint(val));
    if (old == assumed) break;
  }
}

// ---------------- K0: init ----------------
__global__ void init_kernel(const float* __restrict__ interval,
                            const float* __restrict__ Wt, const float* __restrict__ bt,
                            float* tf, float* om1, float* om23) {
  int c = threadIdx.x;  // 128
  float x_t = 1.0f / logf(interval[0] + 2.7182818284590452f);
  tf[c] = tanhf(x_t * Wt[c] + bt[c]);
  om1[c]  = -FLT_MAX;
  om23[c] = -FLT_MAX;
}

// ---------------- K0b: fp32 -> bf16 hi/lo weight split ----------------
__global__ void split_kernel(const float* __restrict__ src, bf16_t* __restrict__ hi,
                             bf16_t* __restrict__ lo, int n) {
  int i = blockIdx.x * 256 + threadIdx.x;
  if (i < n) {
    float v = src[i];
    __bf16 h = (__bf16)v;
    hi[i] = h;
    lo[i] = (__bf16)(v - (float)h);
  }
}

// ---------------- K1: deterministic compaction (1 wave) ----------------
__global__ void compact_kernel(const int* __restrict__ divided,
                               int* idx23, int* idx1, int* counts) {
  int lane = threadIdx.x;             // 0..31
  const int PER = N_CODES / 32;       // 384
  int base = lane * PER;
  int c23 = 0, c1 = 0;
  for (int r = 0; r < PER; ++r) {
    int i = base + r;
    int d1 = divided[i * 3 + 0], d2 = divided[i * 3 + 1], d3 = divided[i * 3 + 2];
    if (d2 | d3) c23++;
    else if (d1) c1++;
  }
  int s23 = c23, s1 = c1;
  for (int off = 1; off < 32; off <<= 1) {
    int t23 = __shfl_up(s23, off, 32);
    int t1  = __shfl_up(s1,  off, 32);
    if (lane >= off) { s23 += t23; s1 += t1; }
  }
  int p23 = s23 - c23, p1 = s1 - c1;  // exclusive prefix
  for (int r = 0; r < PER; ++r) {
    int i = base + r;
    int d1 = divided[i * 3 + 0], d2 = divided[i * 3 + 1], d3 = divided[i * 3 + 2];
    if (d2 | d3) idx23[p23++] = i;
    else if (d1) idx1[p1++] = i;
  }
  if (lane == 31) { counts[0] = s23; counts[1] = s1; }
}

// ---------------- K2: Q/K/V projections on m23 rows (bf16x3 WMMA) ----------------
__global__ __launch_bounds__(32) void proj_kernel(
    const float* __restrict__ co, const float* __restrict__ noe,
    const float* __restrict__ unrel, const int* __restrict__ divided,
    const int* __restrict__ idx23, const int* __restrict__ counts,
    const bf16_t* __restrict__ WqH, const bf16_t* __restrict__ WqL, const float* __restrict__ bq,
    const bf16_t* __restrict__ WkH, const bf16_t* __restrict__ WkL, const float* __restrict__ bk,
    const bf16_t* __restrict__ WvH, const bf16_t* __restrict__ WvL, const float* __restrict__ bv,
    bf16_t* __restrict__ QcH, bf16_t* __restrict__ QcL,
    bf16_t* __restrict__ KcH, bf16_t* __restrict__ KcL,
    bf16_t* __restrict__ VtH, bf16_t* __restrict__ VtL) {
  int Nc = counts[0];
  int j0 = blockIdx.x * 16;
  if (j0 >= Nc) return;                       // wave-uniform, EXEC stays full
  int lane = threadIdx.x, l = lane & 15, h = lane >> 4;

  int jA = j0 + l;
  int iA = idx23[jA < Nc ? jA : (Nc - 1)];
  const float* qrow = (divided[iA * 3 + 1] > 0) ? (noe + (size_t)iA * GRAPH)
                                                : (unrel + (size_t)iA * GRAPH);
  const float* crow = co + (size_t)iA * GRAPH;

  int jd[8];
#pragma unroll
  for (int g = 0; g < 8; ++g) jd[g] = j0 + g + 8 * h;

  // ---- Q (4 col tiles) ----
#pragma unroll 1
  for (int t = 0; t < 4; ++t) {
    int col = 16 * t + l;
    const bf16_t* wh = WqH + (size_t)col * GRAPH;
    const bf16_t* wl = WqL + (size_t)col * GRAPH;
    v8f acc = {};
#pragma unroll
    for (int kk = 0; kk < 4; ++kk) {
      v16bf ah, al;
      loadA_split(qrow, 32 * kk, h, &ah, &al);
      v16bf bh = loadA_bf(wh + 32 * kk, 8 * h, 16 + 8 * h);
      v16bf bl = loadA_bf(wl + 32 * kk, 8 * h, 16 + 8 * h);
      acc = mma3(ah, al, bh, bl, acc);
    }
    float bias = bq[col];
#pragma unroll
    for (int g = 0; g < 8; ++g)
      if (jd[g] < Nc) {
        float v = acc[g] + bias;
        __bf16 hi = (__bf16)v;
        QcH[(size_t)jd[g] * ATT + col] = hi;
        QcL[(size_t)jd[g] * ATT + col] = (__bf16)(v - (float)hi);
      }
  }
  // ---- K (4 col tiles) ----
#pragma unroll 1
  for (int t = 0; t < 4; ++t) {
    int col = 16 * t + l;
    const bf16_t* wh = WkH + (size_t)col * GRAPH;
    const bf16_t* wl = WkL + (size_t)col * GRAPH;
    v8f acc = {};
#pragma unroll
    for (int kk = 0; kk < 4; ++kk) {
      v16bf ah, al;
      loadA_split(crow, 32 * kk, h, &ah, &al);
      v16bf bh = loadA_bf(wh + 32 * kk, 8 * h, 16 + 8 * h);
      v16bf bl = loadA_bf(wl + 32 * kk, 8 * h, 16 + 8 * h);
      acc = mma3(ah, al, bh, bl, acc);
    }
    float bias = bk[col];
#pragma unroll
    for (int g = 0; g < 8; ++g)
      if (jd[g] < Nc) {
        float v = acc[g] + bias;
        __bf16 hi = (__bf16)v;
        KcH[(size_t)jd[g] * ATT + col] = hi;
        KcL[(size_t)jd[g] * ATT + col] = (__bf16)(v - (float)hi);
      }
  }
  // ---- V (8 col tiles), stored transposed Vt[col][j] ----
#pragma unroll 1
  for (int t = 0; t < 8; ++t) {
    int col = 16 * t + l;
    const bf16_t* wh = WvH + (size_t)col * GRAPH;
    const bf16_t* wl = WvL + (size_t)col * GRAPH;
    v8f acc = {};
#pragma unroll
    for (int kk = 0; kk < 4; ++kk) {
      v16bf ah, al;
      loadA_split(crow, 32 * kk, h, &ah, &al);
      v16bf bh = loadA_bf(wh + 32 * kk, 8 * h, 16 + 8 * h);
      v16bf bl = loadA_bf(wl + 32 * kk, 8 * h, 16 + 8 * h);
      acc = mma3(ah, al, bh, bl, acc);
    }
    float bias = bv[col];
#pragma unroll
    for (int g = 0; g < 8; ++g)
      if (jd[g] < Nc) {
        float v = acc[g] + bias;
        __bf16 hi = (__bf16)v;
        VtH[(size_t)col * N_CODES + jd[g]] = hi;
        VtL[(size_t)col * N_CODES + jd[g]] = (__bf16)(v - (float)hi);
      }
  }
}

// ---------------- K3: GRU on compacted m1 rows (bf16x3 WMMA) ----------------
__global__ __launch_bounds__(32) void gru_kernel(
    const float* __restrict__ co, const float* __restrict__ hs,
    const int* __restrict__ idx1, const int* __restrict__ counts,
    const bf16_t* __restrict__ WihH, const bf16_t* __restrict__ WihL,
    const bf16_t* __restrict__ WhhH, const bf16_t* __restrict__ WhhL,
    const float* __restrict__ b_ih, const float* __restrict__ b_hh,
    float* __restrict__ h_new, float* __restrict__ om1) {
  int N1 = counts[1];
  int j0 = blockIdx.x * 16;
  if (j0 >= N1) return;
  int lane = threadIdx.x, l = lane & 15, h = lane >> 4;

  int jA = j0 + l;
  int iA = idx1[jA < N1 ? jA : (N1 - 1)];
  const float* xrow = co + (size_t)iA * GRAPH;
  const float* hrow = hs + (size_t)iA * HIDDEN;

  int jd[8], rid[8];
#pragma unroll
  for (int g = 0; g < 8; ++g) {
    jd[g] = j0 + g + 8 * h;
    rid[g] = idx1[jd[g] < N1 ? jd[g] : (N1 - 1)];
  }

#pragma unroll 1
  for (int t = 0; t < 8; ++t) {
    int col = 16 * t + l;
    const bf16_t* wirH = WihH + (size_t)(col)       * GRAPH;
    const bf16_t* wizH = WihH + (size_t)(128 + col) * GRAPH;
    const bf16_t* winH = WihH + (size_t)(256 + col) * GRAPH;
    const bf16_t* wirL = WihL + (size_t)(col)       * GRAPH;
    const bf16_t* wizL = WihL + (size_t)(128 + col) * GRAPH;
    const bf16_t* winL = WihL + (size_t)(256 + col) * GRAPH;
    const bf16_t* whrH = WhhH + (size_t)(col)       * HIDDEN;
    const bf16_t* whzH = WhhH + (size_t)(128 + col) * HIDDEN;
    const bf16_t* whnH = WhhH + (size_t)(256 + col) * HIDDEN;
    const bf16_t* whrL = WhhL + (size_t)(col)       * HIDDEN;
    const bf16_t* whzL = WhhL + (size_t)(128 + col) * HIDDEN;
    const bf16_t* whnL = WhhL + (size_t)(256 + col) * HIDDEN;
    v8f ir = {}, hr = {}, iz = {}, hz = {}, inn = {}, hn = {};
#pragma unroll 1
    for (int kk = 0; kk < 4; ++kk) {
      int kb = 32 * kk;
      v16bf xh, xl, hh, hl;
      loadA_split(xrow, kb, h, &xh, &xl);
      loadA_split(hrow, kb, h, &hh, &hl);
      v16bf bh, bl;
      bh = loadA_bf(wirH + kb, 8 * h, 16 + 8 * h); bl = loadA_bf(wirL + kb, 8 * h, 16 + 8 * h);
      ir = mma3(xh, xl, bh, bl, ir);
      bh = loadA_bf(whrH + kb, 8 * h, 16 + 8 * h); bl = loadA_bf(whrL + kb, 8 * h, 16 + 8 * h);
      hr = mma3(hh, hl, bh, bl, hr);
      bh = loadA_bf(wizH + kb, 8 * h, 16 + 8 * h); bl = loadA_bf(wizL + kb, 8 * h, 16 + 8 * h);
      iz = mma3(xh, xl, bh, bl, iz);
      bh = loadA_bf(whzH + kb, 8 * h, 16 + 8 * h); bl = loadA_bf(whzL + kb, 8 * h, 16 + 8 * h);
      hz = mma3(hh, hl, bh, bl, hz);
      bh = loadA_bf(winH + kb, 8 * h, 16 + 8 * h); bl = loadA_bf(winL + kb, 8 * h, 16 + 8 * h);
      inn = mma3(xh, xl, bh, bl, inn);
      bh = loadA_bf(whnH + kb, 8 * h, 16 + 8 * h); bl = loadA_bf(whnL + kb, 8 * h, 16 + 8 * h);
      hn = mma3(hh, hl, bh, bl, hn);
    }
    float bir = b_ih[col], biz = b_ih[128 + col], bin_ = b_ih[256 + col];
    float bhr = b_hh[col], bhz = b_hh[128 + col], bhn  = b_hh[256 + col];
    float cm = -FLT_MAX;
#pragma unroll
    for (int g = 0; g < 8; ++g) {
      float r = 1.0f / (1.0f + expf(-(ir[g] + bir + hr[g] + bhr)));
      float z = 1.0f / (1.0f + expf(-(iz[g] + biz + hz[g] + bhz)));
      float n = tanhf(inn[g] + bin_ + r * (hn[g] + bhn));
      float hold = hs[(size_t)rid[g] * HIDDEN + col];
      float out = (1.0f - z) * n + z * hold;
      if (jd[g] < N1) {
        h_new[(size_t)rid[g] * HIDDEN + col] = out;
        cm = fmaxf(cm, out);
      }
    }
    cm = fmaxf(cm, __shfl_xor(cm, 16, 32));
    if (h == 0) atomicMaxF(om1 + col, cm);
  }
}

// ---------------- K4: flash attention over m23 subset (bf16x3) ----------------
__global__ __launch_bounds__(256) void attn_kernel(
    const bf16_t* __restrict__ QcH, const bf16_t* __restrict__ QcL,
    const bf16_t* __restrict__ KcH, const bf16_t* __restrict__ KcL,
    const bf16_t* __restrict__ VtH, const bf16_t* __restrict__ VtL,
    const int* __restrict__ idx23, const int* __restrict__ counts,
    float* __restrict__ h_new, float* __restrict__ om23) {
  __shared__ __align__(16) bf16_t PldsH[8][16][32];   // per-wave P hi (C->A layout bounce)
  __shared__ __align__(16) bf16_t PldsL[8][16][32];   // per-wave P lo
  int Nc = counts[0];
  int tid = threadIdx.x;
  int wave = tid >> 5, lane = tid & 31, l = lane & 15, h = lane >> 4;
  int j0 = blockIdx.x * 128 + wave * 16;
  if (j0 >= Nc) return;                               // wave-uniform

  const bf16_t* qrh = QcH + (size_t)(j0 + l) * ATT;
  const bf16_t* qrl = QcL + (size_t)(j0 + l) * ATT;
  v16bf qh0 = loadA_bf(qrh,      8 * h, 16 + 8 * h);
  v16bf qh1 = loadA_bf(qrh + 32, 8 * h, 16 + 8 * h);
  v16bf ql0 = loadA_bf(qrl,      8 * h, 16 + 8 * h);
  v16bf ql1 = loadA_bf(qrl + 32, 8 * h, 16 + 8 * h);

  v8f o[8];
  float mrun[8], lrun[8];
#pragma unroll
  for (int c = 0; c < 8; ++c) o[c] = {};
#pragma unroll
  for (int g = 0; g < 8; ++g) { mrun[g] = -3.0e38f; lrun[g] = 0.0f; }

  int nkb = (Nc + 31) >> 5;
  for (int kb = 0; kb < nkb; ++kb) {
    int kbase = kb * 32;
    if (kb + 1 < nkb) {  // gfx1250 global_prefetch for next key block
      __builtin_prefetch(KcH + (size_t)(kbase + 32 + l) * ATT, 0, 0);
      __builtin_prefetch(KcL + (size_t)(kbase + 32 + l) * ATT, 0, 0);
    }
    // ---- scores, 2 sub-tiles of 16 keys, bf16x3 over two K=32 chunks ----
    v8f s0 = {}, s1 = {};
    {
      const bf16_t* krH = KcH + (size_t)(kbase + l) * ATT;
      const bf16_t* krL = KcL + (size_t)(kbase + l) * ATT;
      v16bf bh = loadA_bf(krH, 8 * h, 16 + 8 * h);
      v16bf bl = loadA_bf(krL, 8 * h, 16 + 8 * h);
      s0 = mma3(qh0, ql0, bh, bl, s0);
      bh = loadA_bf(krH + 32, 8 * h, 16 + 8 * h);
      bl = loadA_bf(krL + 32, 8 * h, 16 + 8 * h);
      s0 = mma3(qh1, ql1, bh, bl, s0);
    }
    {
      const bf16_t* krH = KcH + (size_t)(kbase + 16 + l) * ATT;
      const bf16_t* krL = KcL + (size_t)(kbase + 16 + l) * ATT;
      v16bf bh = loadA_bf(krH, 8 * h, 16 + 8 * h);
      v16bf bl = loadA_bf(krL, 8 * h, 16 + 8 * h);
      s1 = mma3(qh0, ql0, bh, bl, s1);
      bh = loadA_bf(krH + 32, 8 * h, 16 + 8 * h);
      bl = loadA_bf(krL + 32, 8 * h, 16 + 8 * h);
      s1 = mma3(qh1, ql1, bh, bl, s1);
    }

    bool v0 = (kbase + l) < Nc, v1 = (kbase + 16 + l) < Nc;
#pragma unroll
    for (int g = 0; g < 8; ++g) {
      float sa = v0 ? s0[g] * 0.125f : -1.0e30f;   // 1/sqrt(64)
      float sb = v1 ? s1[g] * 0.125f : -1.0e30f;
      float t = fmaxf(sa, sb);
      t = fmaxf(t, __shfl_xor(t, 1, 32));
      t = fmaxf(t, __shfl_xor(t, 2, 32));
      t = fmaxf(t, __shfl_xor(t, 4, 32));
      t = fmaxf(t, __shfl_xor(t, 8, 32));
      float mnew = fmaxf(mrun[g], t);
      float alpha = expf(mrun[g] - mnew);
      float p0 = expf(sa - mnew);
      float p1 = expf(sb - mnew);
      float rs = p0 + p1;
      rs += __shfl_xor(rs, 1, 32);
      rs += __shfl_xor(rs, 2, 32);
      rs += __shfl_xor(rs, 4, 32);
      rs += __shfl_xor(rs, 8, 32);
      lrun[g] = lrun[g] * alpha + rs;
      mrun[g] = mnew;
#pragma unroll
      for (int c = 0; c < 8; ++c) o[c][g] *= alpha;
      // stash P hi/lo in C-layout: row = g+8h, keys (l, 16+l)
      __bf16 p0h = (__bf16)p0, p1h = (__bf16)p1;
      PldsH[wave][g + 8 * h][l]      = p0h;
      PldsH[wave][g + 8 * h][16 + l] = p1h;
      PldsL[wave][g + 8 * h][l]      = (__bf16)(p0 - (float)p0h);
      PldsL[wave][g + 8 * h][16 + l] = (__bf16)(p1 - (float)p1h);
    }
    asm volatile("s_wait_dscnt 0" ::: "memory");    // per-wave LDS RAW fence
    v16bf pah = loadA_bf(&PldsH[wave][l][0], 8 * h, 16 + 8 * h);
    v16bf pal = loadA_bf(&PldsL[wave][l][0], 8 * h, 16 + 8 * h);
    // ---- PV: 8 output chunks, B^T rows = Vt rows (contiguous over keys) ----
#pragma unroll
    for (int c = 0; c < 8; ++c) {
      const bf16_t* vrh = VtH + (size_t)(16 * c + l) * N_CODES + kbase;
      const bf16_t* vrl = VtL + (size_t)(16 * c + l) * N_CODES + kbase;
      v16bf vbh = loadA_bf(vrh, 8 * h, 16 + 8 * h);
      v16bf vbl = loadA_bf(vrl, 8 * h, 16 + 8 * h);
      o[c] = mma3(pah, pal, vbh, vbl, o[c]);
    }
  }

  // epilogue: normalize, tanh, scatter, column-max
  float cmx[8];
#pragma unroll
  for (int c = 0; c < 8; ++c) cmx[c] = -FLT_MAX;
#pragma unroll
  for (int g = 0; g < 8; ++g) {
    int j = j0 + g + 8 * h;
    if (j < Nc) {
      int row = idx23[j];
      float inv = 1.0f / lrun[g];
#pragma unroll
      for (int c = 0; c < 8; ++c) {
        float val = tanhf(o[c][g] * inv);
        h_new[(size_t)row * HIDDEN + 16 * c + l] = val;
        cmx[c] = fmaxf(cmx[c], val);
      }
    }
  }
#pragma unroll
  for (int c = 0; c < 8; ++c) {
    float cm = fmaxf(cmx[c], __shfl_xor(cmx[c], 16, 32));
    if (h == 0) atomicMaxF(om23 + 16 * c + l, cm);
  }
}

// ---------------- K5: combine ----------------
__global__ void final_kernel(const float* __restrict__ tf, const float* __restrict__ om1,
                             const float* __restrict__ om23, float* __restrict__ out) {
  int c = threadIdx.x;  // 128
  out[c] = fmaxf(om1[c], om23[c]) + tf[c];
}

// ---------------- launcher ----------------
extern "C" void kernel_launch(void* const* d_in, const int* in_sizes, int n_in,
                              void* d_out, int out_size, void* d_ws, size_t ws_size,
                              hipStream_t stream) {
  const float* interval = (const float*)d_in[0];
  const float* co       = (const float*)d_in[1];
  const float* noe      = (const float*)d_in[2];
  const float* unrel    = (const float*)d_in[3];
  const float* hs       = (const float*)d_in[4];
  const int*   divided  = (const int*)  d_in[5];
  const float* W_ih     = (const float*)d_in[6];
  const float* W_hh     = (const float*)d_in[7];
  const float* b_ih     = (const float*)d_in[8];
  const float* b_hh     = (const float*)d_in[9];
  const float* Wq       = (const float*)d_in[10];
  const float* bq       = (const float*)d_in[11];
  const float* Wk       = (const float*)d_in[12];
  const float* bk       = (const float*)d_in[13];
  const float* Wv       = (const float*)d_in[14];
  const float* bv       = (const float*)d_in[15];
  const float* Wt       = (const float*)d_in[16];
  const float* bt       = (const float*)d_in[17];
  (void)in_sizes; (void)n_in; (void)out_size; (void)ws_size;

  char* ws = (char*)d_ws;
  int*    counts = (int*)  (ws + OFF_CNT);
  float*  tf     = (float*)(ws + OFF_TF);
  float*  om1    = (float*)(ws + OFF_M1);
  float*  om23   = (float*)(ws + OFF_M23);
  int*    idx23  = (int*)  (ws + OFF_IDX23);
  int*    idx1   = (int*)  (ws + OFF_IDX1);
  bf16_t* WqH = (bf16_t*)(ws + OFF_WQH), *WqL = (bf16_t*)(ws + OFF_WQL);
  bf16_t* WkH = (bf16_t*)(ws + OFF_WKH), *WkL = (bf16_t*)(ws + OFF_WKL);
  bf16_t* WvH = (bf16_t*)(ws + OFF_WVH), *WvL = (bf16_t*)(ws + OFF_WVL);
  bf16_t* WihH = (bf16_t*)(ws + OFF_WIH_H), *WihL = (bf16_t*)(ws + OFF_WIH_L);
  bf16_t* WhhH = (bf16_t*)(ws + OFF_WHH_H), *WhhL = (bf16_t*)(ws + OFF_WHH_L);
  bf16_t* QcH = (bf16_t*)(ws + OFF_QCH), *QcL = (bf16_t*)(ws + OFF_QCL);
  bf16_t* KcH = (bf16_t*)(ws + OFF_KCH), *KcL = (bf16_t*)(ws + OFF_KCL);
  bf16_t* VtH = (bf16_t*)(ws + OFF_VTH), *VtL = (bf16_t*)(ws + OFF_VTL);

  float* out   = (float*)d_out;
  float* h_new = out + HIDDEN;   // tuple: (output[128], h_new[N,128]) flattened

  hipMemsetAsync(h_new, 0, (size_t)N_CODES * HIDDEN * sizeof(float), stream);
  hipMemsetAsync(ws + OFF_QCH, 0, OFF_END - OFF_QCH, stream);

  init_kernel<<<1, 128, 0, stream>>>(interval, Wt, bt, tf, om1, om23);
  // weight splits (fp32 -> bf16 hi/lo), done once per launch
  split_kernel<<<(ATT * GRAPH + 255) / 256, 256, 0, stream>>>(Wq, WqH, WqL, ATT * GRAPH);
  split_kernel<<<(ATT * GRAPH + 255) / 256, 256, 0, stream>>>(Wk, WkH, WkL, ATT * GRAPH);
  split_kernel<<<(HIDDEN * GRAPH + 255) / 256, 256, 0, stream>>>(Wv, WvH, WvL, HIDDEN * GRAPH);
  split_kernel<<<(3 * HIDDEN * GRAPH + 255) / 256, 256, 0, stream>>>(W_ih, WihH, WihL, 3 * HIDDEN * GRAPH);
  split_kernel<<<(3 * HIDDEN * GRAPH + 255) / 256, 256, 0, stream>>>(W_hh, WhhH, WhhL, 3 * HIDDEN * GRAPH);
  compact_kernel<<<1, 32, 0, stream>>>(divided, idx23, idx1, counts);
  proj_kernel<<<N_CODES / 16, 32, 0, stream>>>(co, noe, unrel, divided, idx23, counts,
                                               WqH, WqL, bq, WkH, WkL, bk, WvH, WvL, bv,
                                               QcH, QcL, KcH, KcL, VtH, VtL);
  gru_kernel<<<N_CODES / 16, 32, 0, stream>>>(co, hs, idx1, counts,
                                              WihH, WihL, WhhH, WhhL, b_ih, b_hh, h_new, om1);
  attn_kernel<<<N_CODES / 128, 256, 0, stream>>>(QcH, QcL, KcH, KcL, VtH, VtL,
                                                 idx23, counts, h_new, om23);
  final_kernel<<<1, 128, 0, stream>>>(tf, om1, om23, out);
}
